// DecentralizedConv_28106265985636
// MI455X (gfx1250) — matrix operand order repaired and checked by reading, not compiled, not run on hardware
//
#include <hip/hip_runtime.h>
#include <hip/hip_bf16.h>

typedef __attribute__((ext_vector_type(16))) _Float16 v16h;
typedef __attribute__((ext_vector_type(8)))  _Float16 v8h;
typedef __attribute__((ext_vector_type(4)))  _Float16 v4h;
typedef __attribute__((ext_vector_type(8)))  float    v8f;

#define DIM 128

// ---------------------------------------------------------------------------
// Zero-fill (accumulators must be zeroed every call; harness does not do it)
// ---------------------------------------------------------------------------
__global__ void zero_f32(float* __restrict__ p, size_t n) {
    size_t i = (size_t)blockIdx.x * blockDim.x + threadIdx.x;
    size_t stride = (size_t)gridDim.x * blockDim.x;
    for (; i < n; i += stride) p[i] = 0.0f;
}

// ---------------------------------------------------------------------------
// LayerNorm over rows of [M,128]; one wave per row; writes f32 and f16 copies
// ---------------------------------------------------------------------------
__global__ void ln_kernel(const float* __restrict__ x,
                          const float* __restrict__ g,
                          const float* __restrict__ b,
                          float* __restrict__ yf,
                          _Float16* __restrict__ yh, int M) {
    int row  = (int)((blockIdx.x * (size_t)blockDim.x + threadIdx.x) >> 5);
    int lane = threadIdx.x & 31;
    if (row >= M) return;
    float4 v = ((const float4*)(x + (size_t)row * DIM))[lane];
    float s = v.x + v.y + v.z + v.w;
    #pragma unroll
    for (int m = 16; m; m >>= 1) s += __shfl_xor(s, m, 32);
    float mean = s * (1.0f / DIM);
    float dx = v.x - mean, dy = v.y - mean, dz = v.z - mean, dw = v.w - mean;
    float ss = dx*dx + dy*dy + dz*dz + dw*dw;
    #pragma unroll
    for (int m = 16; m; m >>= 1) ss += __shfl_xor(ss, m, 32);
    float r = rsqrtf(ss * (1.0f / DIM) + 1e-6f);
    float4 gg = ((const float4*)g)[lane];
    float4 bb = ((const float4*)b)[lane];
    float4 o;
    o.x = dx * r * gg.x + bb.x;
    o.y = dy * r * gg.y + bb.y;
    o.z = dz * r * gg.z + bb.z;
    o.w = dw * r * gg.w + bb.w;
    ((float4*)(yf + (size_t)row * DIM))[lane] = o;
    if (yh) {
        v4h h; h[0] = (_Float16)o.x; h[1] = (_Float16)o.y;
               h[2] = (_Float16)o.z; h[3] = (_Float16)o.w;
        *(v4h*)(yh + (size_t)row * DIM + lane * 4) = h;
    }
}

// ---------------------------------------------------------------------------
// Pack f32 W[128k][128n] row-major -> f16 Wt[n][k] (column-major, contiguous k)
// ---------------------------------------------------------------------------
__global__ void pack_wt(const float* __restrict__ W, _Float16* __restrict__ Wt) {
    int i = blockIdx.x * blockDim.x + threadIdx.x;   // 0..16383
    int k = i >> 7, n = i & 127;
    Wt[n * DIM + k] = (_Float16)W[k * DIM + n];
}

// ---------------------------------------------------------------------------
// rowsum: out[k] = sum_j W[k][j];  out[128] = sum_j bvec[j]
// ---------------------------------------------------------------------------
__global__ void rowsum_plus_bias(const float* __restrict__ W,
                                 const float* __restrict__ bvec,
                                 float* __restrict__ out) {
    int k = threadIdx.x;                              // 128 threads
    float s = 0.f;
    for (int j = 0; j < DIM; ++j) s += W[k * DIM + j];
    out[k] = s;
    if (k == 0) {
        float bs = 0.f;
        for (int j = 0; j < DIM; ++j) bs += bvec[j];
        out[DIM] = bs;
    }
}

// ---------------------------------------------------------------------------
// WMMA GEMM: C[M,128] = epilogue( A_f16[M,128] @ Bt_f16 + bias )
//   epilogue: C = (addend ? addend + scale*(acc+bias) : scale*(acc+bias))
// 8 waves per block, each wave owns one 16x16 tile: 16 rows x 128 cols / block
// ---------------------------------------------------------------------------
union AFrag { v16h v; struct { v8h lo, hi; } h; };

__global__ void gemm128_wmma(const _Float16* __restrict__ A,
                             const _Float16* __restrict__ Bt,
                             const float* __restrict__ bias,
                             const float* __restrict__ addend,
                             float scale,
                             float* __restrict__ C, int M) {
    int wv   = threadIdx.x >> 5;       // 0..7 : column tile
    int lane = threadIdx.x & 31;
    int hs   = lane >> 4;              // half-select
    int l15  = lane & 15;
    int tile_m = blockIdx.x * 16;
    int tile_n = wv * 16;

    int rowA = tile_m + l15; if (rowA >= M) rowA = M - 1;
    const _Float16* arow = A  + (size_t)rowA * DIM;
    const _Float16* bcol = Bt + (size_t)(tile_n + l15) * DIM;

    v8f acc = {};
    int lo = hs * 8;
    #pragma unroll
    for (int k0 = 0; k0 < DIM; k0 += 32) {
        AFrag a, b;
        // A 16x32: lane half 0 -> K {0..7, 16..23}; half 1 -> K {8..15, 24..31}
        a.h.lo = *(const v8h*)(arow + k0 + lo);
        a.h.hi = *(const v8h*)(arow + k0 + 16 + lo);
        // B 32x16: lane half 0 -> K {0..7, 16..23}; half 1 -> K {8..15, 24..31}
        b.h.lo = *(const v8h*)(bcol + k0 + lo);
        b.h.hi = *(const v8h*)(bcol + k0 + 16 + lo);
        acc = __builtin_amdgcn_wmma_f32_16x16x32_f16(
            /*neg_a=*/false, a.v, /*neg_b=*/false, b.v,
            /*c_mod=*/(short)0, acc, /*reuse_a=*/false, /*reuse_b=*/false);
    }

    int col = tile_n + l15;
    float bv = bias ? bias[col] : 0.0f;
    #pragma unroll
    for (int r = 0; r < 8; ++r) {
        int row = tile_m + r + hs * 8;
        if (row < M) {
            float out = scale * (acc[r] + bv);
            if (addend) out += addend[(size_t)row * DIM + col];
            C[(size_t)row * DIM + col] = out;
        }
    }
}

// ---------------------------------------------------------------------------
// sum = tanh( X_row . wext[0..127] + wext[128] ); one wave per row
// ---------------------------------------------------------------------------
__global__ void matvec_tanh(const float* __restrict__ X,
                            const float* __restrict__ wext,
                            float* __restrict__ out, int M) {
    int row  = (int)((blockIdx.x * (size_t)blockDim.x + threadIdx.x) >> 5);
    int lane = threadIdx.x & 31;
    if (row >= M) return;
    float4 v = ((const float4*)(X + (size_t)row * DIM))[lane];
    float4 w = ((const float4*)wext)[lane];
    float s = v.x*w.x + v.y*w.y + v.z*w.z + v.w*w.w;
    #pragma unroll
    for (int m = 16; m; m >>= 1) s += __shfl_xor(s, m, 32);
    if (lane == 0) out[row] = tanhf(s + wext[DIM]);
}

// ---------------------------------------------------------------------------
// out[i] = tanh( dot(T_i, Q_i) ); one wave per row (T already has bias added)
// ---------------------------------------------------------------------------
__global__ void rowdot_tanh(const float* __restrict__ T,
                            const float* __restrict__ Q,
                            float* __restrict__ out, int M) {
    int row  = (int)((blockIdx.x * (size_t)blockDim.x + threadIdx.x) >> 5);
    int lane = threadIdx.x & 31;
    if (row >= M) return;
    float4 t = ((const float4*)(T + (size_t)row * DIM))[lane];
    float4 q = ((const float4*)(Q + (size_t)row * DIM))[lane];
    float s = t.x*q.x + t.y*q.y + t.z*q.z + t.w*q.w;
    #pragma unroll
    for (int m = 16; m; m >>= 1) s += __shfl_xor(s, m, 32);
    if (lane == 0) out[row] = tanhf(s);
}

// ---------------------------------------------------------------------------
// Edge pass A: segsum[src] += exp(leaky_relu(s1[src]+s2[dst]))
// (logits bounded in [-2,2] by tanh, so max-subtraction is unnecessary)
// ---------------------------------------------------------------------------
__global__ void edge_expsum(const int* __restrict__ src, const int* __restrict__ dst,
                            const float* __restrict__ s1, const float* __restrict__ s2,
                            float* __restrict__ segsum, int nE) {
    int e = blockIdx.x * blockDim.x + threadIdx.x;
    if (e >= nE) return;
    int s = src[e];
    float x = s1[s] + s2[dst[e]];
    x = x > 0.f ? x : 0.2f * x;
    atomicAdd(&segsum[s], __expf(x));
}

// ---------------------------------------------------------------------------
// Edge pass B: acc[src] += (exp(logit)/segsum[src]) * val[dst]; 1 wave / edge
// ---------------------------------------------------------------------------
__global__ void edge_scatter(const int* __restrict__ src, const int* __restrict__ dst,
                             const float* __restrict__ s1, const float* __restrict__ s2,
                             const float* __restrict__ segsum,
                             const float* __restrict__ val,
                             float* __restrict__ acc, int nE) {
    int e    = (int)((blockIdx.x * (size_t)blockDim.x + threadIdx.x) >> 5);
    int lane = threadIdx.x & 31;
    if (e >= nE) return;
    int s = src[e], d = dst[e];
    float x = s1[s] + s2[d];
    x = x > 0.f ? x : 0.2f * x;
    float w = __expf(x) / segsum[s];
    float4 vv = ((const float4*)(val + (size_t)d * DIM))[lane];
    float* ap = acc + (size_t)s * DIM + lane * 4;
    atomicAdd(ap + 0, w * vv.x);
    atomicAdd(ap + 1, w * vv.y);
    atomicAdd(ap + 2, w * vv.z);
    atomicAdd(ap + 3, w * vv.w);
}

// ---------------------------------------------------------------------------
extern "C" void kernel_launch(void* const* d_in, const int* in_sizes, int n_in,
                              void* d_out, int out_size, void* d_ws, size_t ws_size,
                              hipStream_t stream) {
    const float* ent_em = (const float*)d_in[0];
    const float* rel_em = (const float*)d_in[1];
    const int* edge_src = (const int*)d_in[2];
    const int* edge_dst = (const int*)d_in[3];
    const int* er_src   = (const int*)d_in[4];
    const int* er_dst   = (const int*)d_in[5];
    const float* la_ln1_g = (const float*)d_in[6],  *la_ln1_b = (const float*)d_in[7];
    const float* la_w_W   = (const float*)d_in[8],  *la_w_b   = (const float*)d_in[9];
    const float* la_w1_W  = (const float*)d_in[10], *la_w1_b  = (const float*)d_in[11];
    const float* la_w2_W  = (const float*)d_in[12], *la_w2_b  = (const float*)d_in[13];
    const float* la_ln2_g = (const float*)d_in[14], *la_ln2_b = (const float*)d_in[15];
    const float* ea_ln1_g = (const float*)d_in[16], *ea_ln1_b = (const float*)d_in[17];
    const float* ea_w_W   = (const float*)d_in[18], *ea_w_b   = (const float*)d_in[19];
    const float* ea_w1_W  = (const float*)d_in[20], *ea_w1_b  = (const float*)d_in[21];
    const float* ea_w2_W  = (const float*)d_in[22], *ea_w2_b  = (const float*)d_in[23];
    const float* ea_ln2_g = (const float*)d_in[24], *ea_ln2_b = (const float*)d_in[25];
    const float* d1_W     = (const float*)d_in[26], *d1_b     = (const float*)d_in[27];

    const int N  = in_sizes[0] / DIM;
    const int R  = in_sizes[1] / DIM;
    const int E  = in_sizes[2];
    const int E2 = in_sizes[4];
    const size_t NB = (size_t)N * DIM;
    const size_t RB = (size_t)R * DIM;

    // ---- workspace carve-up (aggressively reused) ----
    char* base = (char*)d_ws; size_t off = 0;
    auto ALLOC = [&](size_t bytes) -> char* {
        char* p = base + off; off += (bytes + 255) & ~(size_t)255; return p;
    };
    float*    bufA   = (float*)   ALLOC(NB * 4);     // q   -> T1 -> er_out f32
    float*    bufB   = (float*)   ALLOC(NB * 4);     // value -> e_out f32
    float*    bufC   = (float*)   ALLOC(NB * 4);     // acc -> er_acc
    _Float16* bufH   = (_Float16*)ALLOC(NB * 2);     // q_h -> e_out_h -> er_out_h
    float*    segsum = (float*)   ALLOC((size_t)N * 4);
    float*    sum1   = (float*)   ALLOC((size_t)N * 4);  // reused as er_sum1
    float*    sum2   = (float*)   ALLOC((size_t)N * 4);
    float*    w1s    = (float*)   ALLOC(132 * 4);
    float*    w2s    = (float*)   ALLOC(132 * 4);
    float*    relf   = (float*)   ALLOC(RB * 4);
    _Float16* relh   = (_Float16*)ALLOC(RB * 2);
    float*    denseR = (float*)   ALLOC(RB * 4);
    float*    T2     = (float*)   ALLOC(RB * 4);
    float*    sum2r  = (float*)   ALLOC((size_t)R * 4);
    _Float16* Wla    = (_Float16*)ALLOC(DIM * DIM * 2);
    _Float16* Wea    = (_Float16*)ALLOC(DIM * DIM * 2);
    _Float16* Wea1   = (_Float16*)ALLOC(DIM * DIM * 2);
    _Float16* Wea2   = (_Float16*)ALLOC(DIM * DIM * 2);
    _Float16* Wd1    = (_Float16*)ALLOC(DIM * DIM * 2);

    const int wavesPerBlk = 8, TPB = 256;
    const int gRowsN = (N + wavesPerBlk - 1) / wavesPerBlk;
    const int gRowsR = (R + wavesPerBlk - 1) / wavesPerBlk;
    const int gTileN = (N + 15) / 16;
    const int gTileR = (R + 15) / 16;

    // 0) weight packing + rowsum folding
    pack_wt<<<64, 256, 0, stream>>>(la_w_W,  Wla);
    pack_wt<<<64, 256, 0, stream>>>(ea_w_W,  Wea);
    pack_wt<<<64, 256, 0, stream>>>(ea_w1_W, Wea1);
    pack_wt<<<64, 256, 0, stream>>>(ea_w2_W, Wea2);
    pack_wt<<<64, 256, 0, stream>>>(d1_W,    Wd1);
    rowsum_plus_bias<<<1, 128, 0, stream>>>(la_w1_W, la_w1_b, w1s);
    rowsum_plus_bias<<<1, 128, 0, stream>>>(la_w2_W, la_w2_b, w2s);

    // ---- local attention ----
    zero_f32<<<2048, 256, 0, stream>>>(segsum, (size_t)N);
    zero_f32<<<2048, 256, 0, stream>>>(bufC, NB);
    ln_kernel<<<gRowsN, TPB, 0, stream>>>(ent_em, la_ln1_g, la_ln1_b, bufA, bufH, N);
    gemm128_wmma<<<gTileN, TPB, 0, stream>>>(bufH, Wla, la_w_b, nullptr, 1.0f, bufB, N); // value
    matvec_tanh<<<gRowsN, TPB, 0, stream>>>(bufA, w1s, sum1, N);
    matvec_tanh<<<gRowsN, TPB, 0, stream>>>(bufA, w2s, sum2, N);
    edge_expsum<<<(E + TPB - 1) / TPB, TPB, 0, stream>>>(edge_src, edge_dst, sum1, sum2, segsum, E);
    edge_scatter<<<(E + wavesPerBlk - 1) / wavesPerBlk, TPB, 0, stream>>>(
        edge_src, edge_dst, sum1, sum2, segsum, bufB, bufC, E);
    ln_kernel<<<gRowsN, TPB, 0, stream>>>(bufC, la_ln2_g, la_ln2_b, bufB, bufH, N); // e_out (f32+f16)

    // ---- er attention ----
    ln_kernel<<<gRowsR, TPB, 0, stream>>>(rel_em, ea_ln1_g, ea_ln1_b, relf, relh, R);
    gemm128_wmma<<<gTileR, TPB, 0, stream>>>(relh, Wea,  ea_w_b,  nullptr, 1.0f, denseR, R);
    gemm128_wmma<<<gTileR, TPB, 0, stream>>>(relh, Wea2, ea_w2_b, nullptr, 1.0f, T2, R);
    rowdot_tanh<<<gRowsR, TPB, 0, stream>>>(T2, relf, sum2r, R);
    gemm128_wmma<<<gTileN, TPB, 0, stream>>>(bufH, Wea1, ea_w1_b, nullptr, 1.0f, bufA, N); // T1
    rowdot_tanh<<<gRowsN, TPB, 0, stream>>>(bufA, bufB, sum1, N);                          // er_sum1
    zero_f32<<<2048, 256, 0, stream>>>(segsum, (size_t)N);
    zero_f32<<<2048, 256, 0, stream>>>(bufC, NB);
    edge_expsum<<<(E2 + TPB - 1) / TPB, TPB, 0, stream>>>(er_src, er_dst, sum1, sum2r, segsum, E2);
    edge_scatter<<<(E2 + wavesPerBlk - 1) / wavesPerBlk, TPB, 0, stream>>>(
        er_src, er_dst, sum1, sum2r, segsum, denseR, bufC, E2);
    ln_kernel<<<gRowsN, TPB, 0, stream>>>(bufC, ea_ln2_g, ea_ln2_b, bufA, bufH, N); // er_out (f16 in bufH)

    // ---- final: d_out = e_out + 0.6*(er_out @ d1_W + d1_b) ----
    gemm128_wmma<<<gTileN, TPB, 0, stream>>>(bufH, Wd1, d1_b, bufB, 0.6f, (float*)d_out, N);
}
